// AutoRegressive_231928234722
// MI455X (gfx1250) — matrix operand order repaired and checked by reading, not compiled, not run on hardware
//
#include <hip/hip_runtime.h>
#include <hip/hip_bf16.h>
#include <cstdint>

// ---------------- problem constants ----------------
constexpr int Bc  = 4;
constexpr int TTc = 256;
constexpr int TAc = 768;
constexpr int ELc = 225;
constexpr int Dm  = 1024;
constexpr int Hh  = 16;
constexpr int DFF = 4096;
constexpr int NLc = 6;
constexpr int Vc  = 1024;   // codec vocab (output)
constexpr int Ls  = TTc + ELc + TAc;  // 1249 packed length
constexpr int DH  = Dm / Hh;          // 64
constexpr int Lp  = 1280;             // L padded (multiple of 32)
constexpr int Mr  = Bc * Ls;          // 4996 real rows
constexpr int Mp  = 5024;             // rows padded to multiple of 32
constexpr int BH  = Bc * Hh;          // 64 (b,h) pairs
constexpr float EPSc = 1e-5f;

typedef __attribute__((ext_vector_type(16))) _Float16 v16h;
typedef __attribute__((ext_vector_type(8)))  _Float16 h8v;
typedef __attribute__((ext_vector_type(8)))  float    v8f;

static inline unsigned cdiv(size_t a, size_t b) { return (unsigned)((a + b - 1) / b); }

// ---------------- WMMA fragment loaders (CDNA5 16x16x32 f16 layouts) ----------------
// A 16x32 (MxK), row-major, ld = K:
//   lanes 0-15 : m = lane,    halves 0..7 -> K 0..7,  halves 8..15 -> K 16..23
//   lanes16-31 : m = lane-16, halves 0..7 -> K 8..15, halves 8..15 -> K 24..31
__device__ __forceinline__ v16h load_a_frag(const _Float16* __restrict__ A, int lda,
                                            int row0, int k0, int lane) {
  const _Float16* p = A + (size_t)(row0 + (lane & 15)) * lda + (k0 + ((lane >> 4) << 3));
  h8v lo = *(const h8v*)p;
  h8v hi = *(const h8v*)(p + 16);
  v16h r;
#pragma unroll
  for (int i = 0; i < 8; ++i) { r[i] = lo[i]; r[i + 8] = hi[i]; }
  return r;
}

// A fragment from an f32 matrix (attention probabilities), converted on load.
__device__ __forceinline__ v16h load_a_frag_f32(const float* __restrict__ A, int lda,
                                                int row0, int k0, int lane) {
  const float* p = A + (size_t)(row0 + (lane & 15)) * lda + (k0 + ((lane >> 4) << 3));
  float4 a0 = *(const float4*)(p);
  float4 a1 = *(const float4*)(p + 4);
  float4 b0 = *(const float4*)(p + 16);
  float4 b1 = *(const float4*)(p + 20);
  v16h r;
  r[0]  = (_Float16)a0.x; r[1]  = (_Float16)a0.y; r[2]  = (_Float16)a0.z; r[3]  = (_Float16)a0.w;
  r[4]  = (_Float16)a1.x; r[5]  = (_Float16)a1.y; r[6]  = (_Float16)a1.z; r[7]  = (_Float16)a1.w;
  r[8]  = (_Float16)b0.x; r[9]  = (_Float16)b0.y; r[10] = (_Float16)b0.z; r[11] = (_Float16)b0.w;
  r[12] = (_Float16)b1.x; r[13] = (_Float16)b1.y; r[14] = (_Float16)b1.z; r[15] = (_Float16)b1.w;
  return r;
}

// B 32x16 (KxN). We compute C = A * W^T with W stored [N,K] row-major, so
// B[k][n] = W[n][k]: per lane n = lane&15 fixed, same k order as A.
//   lanes 0-15: K 0..7,16..23 ; lanes 16-31: K 8..15,24..31
__device__ __forceinline__ v16h load_b_frag(const _Float16* __restrict__ W, int ldw,
                                            int col0, int k0, int lane) {
  const _Float16* p = W + (size_t)(col0 + (lane & 15)) * ldw + (k0 + ((lane >> 4) << 3));
  h8v lo = *(const h8v*)p;
  h8v hi = *(const h8v*)(p + 16);
  v16h r;
#pragma unroll
  for (int i = 0; i < 8; ++i) { r[i] = lo[i]; r[i + 8] = hi[i]; }
  return r;
}

// ---------------- generic batched WMMA GEMM: C = scale*(A @ W^T) + bias ----------------
// A: [M,K] f16 row-major, W: [N,K] f16 row-major, C: [M,N] f32.
// One wave computes a 32(M) x 64(N) tile: 2 A-fragments x 4 B-fragments -> 8 WMMAs/k-step.
// AI ~= 2*32*64 / ((32+64)*2B) = 21 flop/byte streamed, vs 12.8 for a 16x64 strip.
__global__ void gemm_f16(const _Float16* __restrict__ A, long long sA,
                         const _Float16* __restrict__ W, long long sW,
                         const float* __restrict__ bias,
                         float* __restrict__ C, long long sC,
                         int M, int N, int K, float scale) {
  A += (size_t)blockIdx.y * sA;
  W += (size_t)blockIdx.y * sW;
  C += (size_t)blockIdx.y * sC;
  int wave = blockIdx.x * (blockDim.x >> 5) + (threadIdx.x >> 5);
  int lane = threadIdx.x & 31;
  int stripsN = N >> 6;
  int tm = wave / stripsN;
  int tn = wave - tm * stripsN;
  if (tm * 32 >= M) return;                 // wave-uniform: EXEC stays all-ones
  int row0 = tm << 5, col0 = tn << 6;
  v8f zero{};
  v8f acc[2][4] = {{zero, zero, zero, zero}, {zero, zero, zero, zero}};
  for (int k0 = 0; k0 < K; k0 += 32) {
    v16h a0 = load_a_frag(A, K, row0,      k0, lane);
    v16h a1 = load_a_frag(A, K, row0 + 16, k0, lane);
    v16h b0 = load_b_frag(W, K, col0,      k0, lane);
    v16h b1 = load_b_frag(W, K, col0 + 16, k0, lane);
    v16h b2 = load_b_frag(W, K, col0 + 32, k0, lane);
    v16h b3 = load_b_frag(W, K, col0 + 48, k0, lane);
    acc[0][0] = __builtin_amdgcn_wmma_f32_16x16x32_f16(false, a0, false, b0, (short)0, acc[0][0], false, false);
    acc[0][1] = __builtin_amdgcn_wmma_f32_16x16x32_f16(false, a0, false, b1, (short)0, acc[0][1], false, false);
    acc[0][2] = __builtin_amdgcn_wmma_f32_16x16x32_f16(false, a0, false, b2, (short)0, acc[0][2], false, false);
    acc[0][3] = __builtin_amdgcn_wmma_f32_16x16x32_f16(false, a0, false, b3, (short)0, acc[0][3], false, false);
    acc[1][0] = __builtin_amdgcn_wmma_f32_16x16x32_f16(false, a1, false, b0, (short)0, acc[1][0], false, false);
    acc[1][1] = __builtin_amdgcn_wmma_f32_16x16x32_f16(false, a1, false, b1, (short)0, acc[1][1], false, false);
    acc[1][2] = __builtin_amdgcn_wmma_f32_16x16x32_f16(false, a1, false, b2, (short)0, acc[1][2], false, false);
    acc[1][3] = __builtin_amdgcn_wmma_f32_16x16x32_f16(false, a1, false, b3, (short)0, acc[1][3], false, false);
  }
  // C/D layout: lanes 0-15 -> N=lane, M=+i ; lanes 16-31 -> N=lane-16, M=+8+i
  int m0 = (lane >> 4) << 3;
  int nb = col0 + (lane & 15);
#pragma unroll
  for (int rb = 0; rb < 2; ++rb) {
    int rowb = row0 + rb * 16 + m0;
#pragma unroll
    for (int t = 0; t < 4; ++t) {
      int n = nb + t * 16;
      float bv = bias ? bias[n] : 0.0f;
#pragma unroll
      for (int i = 0; i < 8; ++i)
        C[(size_t)(rowb + i) * N + n] = acc[rb][t][i] * scale + bv;
    }
  }
}

// P(f32) @ V variant: A is the f32 softmax output, B is V^T ([DH, Lp] f16).
__global__ void gemm_pv(const float* __restrict__ A, long long sA,
                        const _Float16* __restrict__ W, long long sW,
                        float* __restrict__ C, long long sC,
                        int M, int N, int K) {
  A += (size_t)blockIdx.y * sA;
  W += (size_t)blockIdx.y * sW;
  C += (size_t)blockIdx.y * sC;
  int wave = blockIdx.x * (blockDim.x >> 5) + (threadIdx.x >> 5);
  int lane = threadIdx.x & 31;
  int stripsN = N >> 6;
  int tm = wave / stripsN;
  int tn = wave - tm * stripsN;
  if (tm * 32 >= M) return;
  int row0 = tm << 5, col0 = tn << 6;
  v8f zero{};
  v8f acc[2][4] = {{zero, zero, zero, zero}, {zero, zero, zero, zero}};
  for (int k0 = 0; k0 < K; k0 += 32) {
    v16h a0 = load_a_frag_f32(A, K, row0,      k0, lane);
    v16h a1 = load_a_frag_f32(A, K, row0 + 16, k0, lane);
    v16h b0 = load_b_frag(W, K, col0,      k0, lane);
    v16h b1 = load_b_frag(W, K, col0 + 16, k0, lane);
    v16h b2 = load_b_frag(W, K, col0 + 32, k0, lane);
    v16h b3 = load_b_frag(W, K, col0 + 48, k0, lane);
    acc[0][0] = __builtin_amdgcn_wmma_f32_16x16x32_f16(false, a0, false, b0, (short)0, acc[0][0], false, false);
    acc[0][1] = __builtin_amdgcn_wmma_f32_16x16x32_f16(false, a0, false, b1, (short)0, acc[0][1], false, false);
    acc[0][2] = __builtin_amdgcn_wmma_f32_16x16x32_f16(false, a0, false, b2, (short)0, acc[0][2], false, false);
    acc[0][3] = __builtin_amdgcn_wmma_f32_16x16x32_f16(false, a0, false, b3, (short)0, acc[0][3], false, false);
    acc[1][0] = __builtin_amdgcn_wmma_f32_16x16x32_f16(false, a1, false, b0, (short)0, acc[1][0], false, false);
    acc[1][1] = __builtin_amdgcn_wmma_f32_16x16x32_f16(false, a1, false, b1, (short)0, acc[1][1], false, false);
    acc[1][2] = __builtin_amdgcn_wmma_f32_16x16x32_f16(false, a1, false, b2, (short)0, acc[1][2], false, false);
    acc[1][3] = __builtin_amdgcn_wmma_f32_16x16x32_f16(false, a1, false, b3, (short)0, acc[1][3], false, false);
  }
  int m0 = (lane >> 4) << 3;
  int nb = col0 + (lane & 15);
#pragma unroll
  for (int rb = 0; rb < 2; ++rb) {
    int rowb = row0 + rb * 16 + m0;
#pragma unroll
    for (int t = 0; t < 4; ++t) {
      int n = nb + t * 16;
#pragma unroll
      for (int i = 0; i < 8; ++i)
        C[(size_t)(rowb + i) * N + n] = acc[rb][t][i];
    }
  }
}

// ---------------- elementwise / support kernels ----------------
__global__ void f32_to_f16_k(const float* __restrict__ s, _Float16* __restrict__ d, int n) {
  int i = blockIdx.x * 256 + threadIdx.x;
  if (i < n) d[i] = (_Float16)s[i];
}

__global__ void relu_to_f16_k(const float* __restrict__ s, _Float16* __restrict__ d, int n) {
  int i = blockIdx.x * 256 + threadIdx.x;
  if (i < n) d[i] = (_Float16)fmaxf(s[i], 0.0f);
}

// embedding + ragged pack + sinusoid
__global__ void embed_k(const int* __restrict__ text, const int* __restrict__ audio,
                        const int* __restrict__ enrolled,
                        const int* __restrict__ tlen, const int* __restrict__ alen,
                        const float* __restrict__ text_emb, const float* __restrict__ audio_emb,
                        float* __restrict__ x, _Float16* __restrict__ xh,
                        _Float16* __restrict__ memh) {
  int idx = blockIdx.x * 256 + threadIdx.x;
  if (idx >= Mr * Dm) return;
  int d = idx & (Dm - 1);
  int rp = idx >> 10;            // b*Ls + p
  int b = rp / Ls;
  int p = rp - b * Ls;
  int tl = tlen[b], al = alen[b];
  float val = 0.0f;
  int pos = -1, tok = 0;
  const float* emb = nullptr;
  if (p < tl)                { pos = p;            tok = text[b * TTc + pos];     emb = text_emb;  }
  else if (p < tl + ELc)     { pos = p - tl;       tok = enrolled[b * ELc + pos]; emb = audio_emb; }
  else if (p < tl + ELc + al){ pos = p - tl - ELc; tok = audio[b * TAc + pos];    emb = audio_emb; }
  if (pos >= 0) {
    int p2 = d >> 1;
    float freq = __expf(-(float)(2 * p2) * (9.210340371976184f / (float)Dm)); // ln(10000)
    float ang = (float)pos * freq;
    float pe = (d & 1) ? __cosf(ang) : __sinf(ang);
    val = emb[(size_t)tok * Dm + d] + pe;
  }
  size_t o = (size_t)rp * Dm + d;
  x[o] = val; xh[o] = (_Float16)val; memh[o] = (_Float16)val;
}

// y [Mp,Dm] f32 -> per-head q/k [BH, Lp, DH] f16 (zero-padded rows)
__global__ void repack_qk_k(const float* __restrict__ y, _Float16* __restrict__ out) {
  int idx = blockIdx.x * 256 + threadIdx.x;
  if (idx >= BH * Lp * DH) return;
  int d  = idx & (DH - 1);
  int p  = (idx / DH) % Lp;
  int bh = idx / (DH * Lp);
  int b = bh / Hh, h = bh - b * Hh;
  float v = 0.0f;
  if (p < Ls) v = y[(size_t)(b * Ls + p) * Dm + h * DH + d];
  out[idx] = (_Float16)v;
}

// y [Mp,Dm] f32 -> transposed per-head V: vt [BH, DH, Lp] f16
__global__ void repack_vt_k(const float* __restrict__ y, _Float16* __restrict__ out) {
  int idx = blockIdx.x * 256 + threadIdx.x;
  if (idx >= BH * Lp * DH) return;
  int d  = idx & (DH - 1);
  int p  = (idx / DH) % Lp;
  int bh = idx / (DH * Lp);
  int b = bh / Hh, h = bh - b * Hh;
  float v = 0.0f;
  if (p < Ls) v = y[(size_t)(b * Ls + p) * Dm + h * DH + d];
  out[((size_t)bh * DH + d) * Lp + p] = (_Float16)v;
}

// O [BH, Lp, DH] f32 -> concat heads oc [Mp, Dm] f16
__global__ void repack_o_k(const float* __restrict__ O, _Float16* __restrict__ oc) {
  int idx = blockIdx.x * 256 + threadIdx.x;
  if (idx >= BH * Ls * DH) return;
  int d  = idx & (DH - 1);
  int p  = (idx / DH) % Ls;
  int bh = idx / (DH * Ls);
  int b = bh / Hh, h = bh - b * Hh;
  oc[(size_t)(b * Ls + p) * Dm + h * DH + d] = (_Float16)O[((size_t)bh * Lp + p) * DH + d];
}

// masked softmax, in place on S [BH, Lp, Lp] f32.
// reference quirk: mask batch index = (b*H + h) % B == bh % B.
__global__ void softmax_k(float* __restrict__ S, const int* __restrict__ tlen,
                          const int* __restrict__ alen, int isSelf) {
  int i  = blockIdx.x;   // query row (0..Lp-1)
  int bh = blockIdx.y;
  int mb = bh % Bc;
  int prefix = tlen[mb] + ELc;
  int item   = prefix + alen[mb];
  float* row = S + ((size_t)bh * Lp + i) * Lp;
  int tid = threadIdx.x;
  __shared__ float red[256];
  float mx = -1e30f;
  for (int j = tid; j < Ls; j += 256) {
    bool neg = isSelf && (j > i) && (j >= prefix) && (i < item) && (j < item);
    if (!neg) mx = fmaxf(mx, row[j]);
  }
  red[tid] = mx; __syncthreads();
  for (int s = 128; s > 0; s >>= 1) { if (tid < s) red[tid] = fmaxf(red[tid], red[tid + s]); __syncthreads(); }
  mx = red[0]; __syncthreads();
  float sum = 0.0f;
  for (int j = tid; j < Lp; j += 256) {
    float e = 0.0f;
    if (j < Ls) {
      bool neg = isSelf && (j > i) && (j >= prefix) && (i < item) && (j < item);
      if (!neg) e = __expf(row[j] - mx);
    }
    row[j] = e; sum += e;
  }
  red[tid] = sum; __syncthreads();
  for (int s = 128; s > 0; s >>= 1) { if (tid < s) red[tid] += red[tid + s]; __syncthreads(); }
  float inv = 1.0f / red[0];
  for (int j = tid; j < Lp; j += 256) row[j] *= inv;
}

// x = LayerNorm(x + add); also refresh f16 shadow. One block per row, D=1024.
__global__ void add_ln_k(float* __restrict__ x, const float* __restrict__ add,
                         const float* __restrict__ g, const float* __restrict__ be,
                         _Float16* __restrict__ xh) {
  int row = blockIdx.x, tid = threadIdx.x;
  float* xr = x + (size_t)row * Dm;
  const float* ar = add + (size_t)row * Dm;
  float t[4]; float s = 0.0f;
#pragma unroll
  for (int i = 0; i < 4; ++i) { t[i] = xr[tid + i * 256] + ar[tid + i * 256]; s += t[i]; }
  __shared__ float red[256];
  red[tid] = s; __syncthreads();
  for (int k = 128; k > 0; k >>= 1) { if (tid < k) red[tid] += red[tid + k]; __syncthreads(); }
  float mean = red[0] * (1.0f / Dm); __syncthreads();
  float vs = 0.0f;
#pragma unroll
  for (int i = 0; i < 4; ++i) { float d0 = t[i] - mean; vs += d0 * d0; }
  red[tid] = vs; __syncthreads();
  for (int k = 128; k > 0; k >>= 1) { if (tid < k) red[tid] += red[tid + k]; __syncthreads(); }
  float inv = rsqrtf(red[0] * (1.0f / Dm) + EPSc);
#pragma unroll
  for (int i = 0; i < 4; ++i) {
    int d0 = tid + i * 256;
    float o = g[d0] * (t[i] - mean) * inv + be[d0];
    xr[d0] = o;
    xh[(size_t)row * Dm + d0] = (_Float16)o;
  }
}

// ---------------- host orchestration ----------------
extern "C" void kernel_launch(void* const* d_in, const int* in_sizes, int n_in,
                              void* d_out, int out_size, void* d_ws, size_t ws_size,
                              hipStream_t stream) {
  (void)in_sizes; (void)n_in; (void)out_size; (void)ws_size;
  // ---- input layout (setup_inputs insertion order, params flattened depth-first) ----
  const int* text     = (const int*)d_in[0];
  const int* audio    = (const int*)d_in[1];
  const int* enrolled = (const int*)d_in[2];
  const int* tlen     = (const int*)d_in[3];
  const int* alen     = (const int*)d_in[4];
  auto F = [&](int i) { return (const float*)d_in[i]; };
  const float* text_emb  = F(5);
  const float* audio_emb = F(6);
  auto LIDX = [](int l, int k) { return 7 + l * 26 + k; };
  // layer k: 0-7 sa{Wq,Wk,Wv,Wo,bq,bk,bv,bo}; 8-15 ca; 16/17 ln1 g,b; 18/19 ln2; 20/21 ln3;
  //          22 W1, 23 b1, 24 W2, 25 b2.  Then 163 W_out, 164 b_out.
  const float* Wout_f = F(7 + NLc * 26);
  const float* b_out  = F(7 + NLc * 26 + 1);

  // ---- workspace bump allocator ----
  uintptr_t base = (uintptr_t)d_ws;
  size_t off = 0;
  auto alloc = [&](size_t bytes) -> void* {
    off = (off + 255) & ~(size_t)255;
    void* p = (void*)(base + off);
    off += bytes;
    return p;
  };
  float*     xf   = (float*)    alloc((size_t)Mp * Dm * 4);
  _Float16*  x_h  = (_Float16*) alloc((size_t)Mp * Dm * 2);
  _Float16*  memh = (_Float16*) alloc((size_t)Mp * Dm * 2);
  float*     yb   = (float*)    alloc((size_t)Mp * DFF * 4);   // large GEMM out (also final logits)
  float*     y2b  = (float*)    alloc((size_t)Mp * Dm * 4);    // residual branch out
  _Float16*  hh   = (_Float16*) alloc((size_t)Mp * DFF * 2);   // relu(f16)
  _Float16*  och  = (_Float16*) alloc((size_t)Mp * Dm * 2);    // concat attn heads
  _Float16*  qh   = (_Float16*) alloc((size_t)BH * Lp * DH * 2);
  _Float16*  kh   = (_Float16*) alloc((size_t)BH * Lp * DH * 2);
  _Float16*  vth  = (_Float16*) alloc((size_t)BH * DH * Lp * 2);
  float*     Sb   = (float*)    alloc((size_t)BH * Lp * Lp * 4);
  float*     Ob   = (float*)    alloc((size_t)BH * Lp * DH * 4);
  _Float16*  w16  = (_Float16*) alloc((size_t)(NLc * (8 * Dm * Dm + 2 * Dm * DFF) + Vc * Dm) * 2);

  hipMemsetAsync(xf,   0, (size_t)Mp * Dm * 4, stream);
  hipMemsetAsync(x_h,  0, (size_t)Mp * Dm * 2, stream);
  hipMemsetAsync(memh, 0, (size_t)Mp * Dm * 2, stream);
  hipMemsetAsync(och,  0, (size_t)Mp * Dm * 2, stream);

  // ---- convert all weight matrices to f16 once ----
  size_t woff = 0;
  auto conv = [&](const float* src, size_t n) -> _Float16* {
    _Float16* dst = w16 + woff; woff += n;
    f32_to_f16_k<<<cdiv(n, 256), 256, 0, stream>>>(src, dst, (int)n);
    return dst;
  };
  _Float16 *Wq[NLc][2], *Wk[NLc][2], *Wv[NLc][2], *Wo[NLc][2], *W1h[NLc], *W2h[NLc];
  for (int l = 0; l < NLc; ++l) {
    for (int a = 0; a < 2; ++a) {
      int bidx = LIDX(l, a * 8);
      Wq[l][a] = conv(F(bidx + 0), (size_t)Dm * Dm);
      Wk[l][a] = conv(F(bidx + 1), (size_t)Dm * Dm);
      Wv[l][a] = conv(F(bidx + 2), (size_t)Dm * Dm);
      Wo[l][a] = conv(F(bidx + 3), (size_t)Dm * Dm);
    }
    W1h[l] = conv(F(LIDX(l, 22)), (size_t)DFF * Dm);
    W2h[l] = conv(F(LIDX(l, 24)), (size_t)Dm * DFF);
  }
  _Float16* WoutH = conv(Wout_f, (size_t)Vc * Dm);

  // ---- embed + pack ----
  embed_k<<<cdiv((size_t)Mr * Dm, 256), 256, 0, stream>>>(
      text, audio, enrolled, tlen, alen, text_emb, audio_emb, xf, x_h, memh);

  // ---- GEMM launcher (32x64 tiles per wave, 8 waves per block) ----
  auto gemm = [&](const _Float16* A, long long sA, const _Float16* W, long long sW,
                  const float* bias, float* C, long long sC,
                  int M, int N, int K, float scale, int batches) {
    int waves = (M / 32) * (N / 64);
    dim3 grid((unsigned)((waves + 7) / 8), (unsigned)batches);
    gemm_f16<<<grid, 256, 0, stream>>>(A, sA, W, sW, bias, C, sC, M, N, K, scale);
  };

  const unsigned g_qk = cdiv((size_t)BH * Lp * DH, 256);
  const unsigned g_o  = cdiv((size_t)BH * Ls * DH, 256);

  auto attention = [&](const _Float16* kvin, int l, int a /*0=sa,1=ca*/, int isSelf) {
    int bidx = LIDX(l, a * 8);
    gemm(x_h, 0, Wq[l][a], 0, F(bidx + 4), yb, 0, Mp, Dm, Dm, 1.0f, 1);
    repack_qk_k<<<g_qk, 256, 0, stream>>>(yb, qh);
    gemm(kvin, 0, Wk[l][a], 0, F(bidx + 5), yb, 0, Mp, Dm, Dm, 1.0f, 1);
    repack_qk_k<<<g_qk, 256, 0, stream>>>(yb, kh);
    gemm(kvin, 0, Wv[l][a], 0, F(bidx + 6), yb, 0, Mp, Dm, Dm, 1.0f, 1);
    repack_vt_k<<<g_qk, 256, 0, stream>>>(yb, vth);
    // S = (Q K^T) / sqrt(DH), batched over (b,h)
    gemm(qh, (long long)Lp * DH, kh, (long long)Lp * DH, nullptr,
         Sb, (long long)Lp * Lp, Lp, Lp, DH, 0.125f, BH);
    softmax_k<<<dim3(Lp, BH), 256, 0, stream>>>(Sb, tlen, alen, isSelf);
    // O = P V
    {
      int waves = (Lp / 32) * (DH / 64);
      dim3 grid((unsigned)((waves + 7) / 8), (unsigned)BH);
      gemm_pv<<<grid, 256, 0, stream>>>(Sb, (long long)Lp * Lp, vth, (long long)DH * Lp,
                                        Ob, (long long)Lp * DH, Lp, DH, Lp);
    }
    repack_o_k<<<g_o, 256, 0, stream>>>(Ob, och);
    gemm(och, 0, Wo[l][a], 0, F(bidx + 7), y2b, 0, Mp, Dm, Dm, 1.0f, 1);
  };

  for (int l = 0; l < NLc; ++l) {
    // self-attention + LN1
    attention(x_h, l, 0, 1);
    add_ln_k<<<Mp, 256, 0, stream>>>(xf, y2b, F(LIDX(l, 16)), F(LIDX(l, 17)), x_h);
    // cross-attention (K/V from encoder memory) + LN2
    attention(memh, l, 1, 0);
    add_ln_k<<<Mp, 256, 0, stream>>>(xf, y2b, F(LIDX(l, 18)), F(LIDX(l, 19)), x_h);
    // FFN + LN3
    gemm(x_h, 0, W1h[l], 0, F(LIDX(l, 23)), yb, 0, Mp, DFF, Dm, 1.0f, 1);
    relu_to_f16_k<<<cdiv((size_t)Mp * DFF, 256), 256, 0, stream>>>(yb, hh, Mp * DFF);
    gemm(hh, 0, W2h[l], 0, F(LIDX(l, 25)), y2b, 0, Mp, Dm, DFF, 1.0f, 1);
    add_ln_k<<<Mp, 256, 0, stream>>>(xf, y2b, F(LIDX(l, 20)), F(LIDX(l, 21)), x_h);
  }

  // final projection to codec vocab, then copy the real rows to d_out ([B,L,V] f32)
  gemm(x_h, 0, WoutH, 0, b_out, yb, 0, Mp, Vc, Dm, 1.0f, 1);
  hipMemcpyAsync(d_out, yb, (size_t)Mr * Vc * sizeof(float),
                 hipMemcpyDeviceToDevice, stream);
}
